// FluxAttention_68865505624070
// MI455X (gfx1250) — matrix operand order repaired and checked by reading, not compiled, not run on hardware
//
#include <hip/hip_runtime.h>

typedef __bf16 bf16;
typedef __attribute__((ext_vector_type(16))) __bf16 v16bf;
typedef __attribute__((ext_vector_type(4)))  __bf16 v4bf;
typedef __attribute__((ext_vector_type(8)))  float  v8f;

struct U128 { unsigned int x, y, z, w; };
struct U128x2 { U128 a, b; };

static __device__ __forceinline__ unsigned short bfbits(float f) {
  unsigned u = __builtin_bit_cast(unsigned, f);
  return (unsigned short)((u + 0x7FFFu + ((u >> 16) & 1u)) >> 16);  // RNE
}
static __device__ __forceinline__ bf16 f2bf(float f) {
  return __builtin_bit_cast(bf16, bfbits(f));
}
static __device__ __forceinline__ unsigned pack2(float a, float b) {
  return (unsigned)bfbits(a) | ((unsigned)bfbits(b) << 16);
}

// Build a 16-element bf16 WMMA fragment from two 16-byte LDS reads.
static __device__ __forceinline__ v16bf load_frag(const bf16* p0, const bf16* p1) {
  U128x2 t;
  t.a = *(const U128*)p0;
  t.b = *(const U128*)p1;
  return __builtin_bit_cast(v16bf, t);
}

// ---------------------------------------------------------------------------
// Generic tiled GEMM: C[M,N] = A[M,K] @ W[K,N] (+bias), f32 in/out, bf16 WMMA.
// Block: 256 threads (8 waves). Tile: 128x128, K-step 32 (one WMMA K).
// Wave grid 2x4 -> each wave owns 64x32 = 4x2 accumulator tiles.
// M,N multiples of 128; K multiple of 32 (always true here).
// ---------------------------------------------------------------------------
template <bool ADD_BIAS>
__global__ __launch_bounds__(256) void gemm_bf16_kernel(
    const float* __restrict__ A, const float* __restrict__ W,
    const float* __restrict__ bias, float* __restrict__ C,
    int M, int N, int K)
{
  __shared__ __align__(16) bf16 As[128][40];   // [m][k], padded 16B-aligned rows
  __shared__ __align__(16) bf16 Bs[128][40];   // [n][k] (W tile transposed)

  const int tid  = threadIdx.x;
  const int lane = tid & 31;
  const int w    = tid >> 5;
  const int wm   = w >> 2;       // 0..1
  const int wn   = w & 3;        // 0..3
  const int half = lane >> 4;    // 0..1
  const int ln   = lane & 15;

  const int rowBase = blockIdx.y * 128;
  const int colBase = blockIdx.x * 128;

  const v8f zero8 = {0.f,0.f,0.f,0.f,0.f,0.f,0.f,0.f};
  v8f acc[4][2];
  #pragma unroll
  for (int i = 0; i < 4; ++i)
    #pragma unroll
    for (int j = 0; j < 2; ++j) acc[i][j] = zero8;

  // A tile: 128x32, 16 contiguous f32 per thread.
  const int arow = tid >> 1;
  const int acol = (tid & 1) * 16;
  // W tile: 32x128; thread owns column n = tid&127, k-range 16*(tid>>7).
  const int bn = tid & 127;
  const int bk = (tid >> 7) * 16;

  for (int k0 = 0; k0 < K; k0 += 32) {
    __syncthreads();
    {
      const float* ap = A + (size_t)(rowBase + arow) * K + k0 + acol;
      float4 f0 = *(const float4*)(ap + 0);
      float4 f1 = *(const float4*)(ap + 4);
      float4 f2 = *(const float4*)(ap + 8);
      float4 f3 = *(const float4*)(ap + 12);
      U128 lo { pack2(f0.x,f0.y), pack2(f0.z,f0.w), pack2(f1.x,f1.y), pack2(f1.z,f1.w) };
      U128 hi { pack2(f2.x,f2.y), pack2(f2.z,f2.w), pack2(f3.x,f3.y), pack2(f3.z,f3.w) };
      *(U128*)&As[arow][acol + 0] = lo;
      *(U128*)&As[arow][acol + 8] = hi;

      const float* wp = W + (size_t)(k0 + bk) * N + colBase + bn;
      float t[16];
      #pragma unroll
      for (int j = 0; j < 16; ++j) t[j] = wp[(size_t)j * N];
      U128 wl { pack2(t[0],t[1]),  pack2(t[2],t[3]),  pack2(t[4],t[5]),  pack2(t[6],t[7]) };
      U128 wh { pack2(t[8],t[9]),  pack2(t[10],t[11]), pack2(t[12],t[13]), pack2(t[14],t[15]) };
      *(U128*)&Bs[bn][bk + 0] = wl;
      *(U128*)&Bs[bn][bk + 8] = wh;

      if (k0 + 32 < K) {  // prefetch next k-slab (global_prefetch_b8)
        __builtin_prefetch(ap + 32, 0, 3);
        __builtin_prefetch(wp + (size_t)32 * N, 0, 3);
      }
    }
    __syncthreads();

    // A fragments: lane(half,m): j -> K = half*16 + j
    v16bf afrag[4];
    #pragma unroll
    for (int mt = 0; mt < 4; ++mt) {
      const bf16* p = &As[wm*64 + mt*16 + ln][0];
      afrag[mt] = load_frag(p + half*16, p + half*16 + 8);
    }
    #pragma unroll
    for (int nt = 0; nt < 2; ++nt) {
      // B fragment: lane(half,n): j -> K = half*16 + j (contiguous in Bs[n][k])
      const bf16* bp = &Bs[wn*32 + nt*16 + ln][half*16];
      v16bf bfrag = load_frag(bp, bp + 8);
      #pragma unroll
      for (int mt = 0; mt < 4; ++mt)
        acc[mt][nt] = __builtin_amdgcn_wmma_f32_16x16x32_bf16(
            false, afrag[mt], false, bfrag, (short)0, acc[mt][nt], false, false);
    }
  }

  // Epilogue. C layout: lane(half,n), VGPR r -> (M = half*8 + r, N = n)
  #pragma unroll
  for (int mt = 0; mt < 4; ++mt) {
    #pragma unroll
    for (int nt = 0; nt < 2; ++nt) {
      const int col = colBase + wn*32 + nt*16 + ln;
      const float b = ADD_BIAS ? bias[col] : 0.f;
      #pragma unroll
      for (int r = 0; r < 8; ++r) {
        const int row = rowBase + wm*64 + mt*16 + half*8 + r;
        C[(size_t)row * N + col] = acc[mt][nt][r] + b;
      }
    }
  }
}

// ---------------------------------------------------------------------------
// Per-(token,head) RMS-norm + RoPE + bf16 pack into [H][S][D].
// One wave per (t,h); lane owns 4 consecutive d (RoPE pairs stay in-lane).
// Q is pre-scaled by 1/sqrt(D).
// ---------------------------------------------------------------------------
__global__ __launch_bounds__(256) void norm_rope_pack_kernel(
    const float* __restrict__ qkv,       // [2560][9216], txt rows first
    const float* __restrict__ gq,  const float* __restrict__ gk,
    const float* __restrict__ gaq, const float* __restrict__ gak,
    const float* __restrict__ rope_cos, const float* __restrict__ rope_sin,
    bf16* __restrict__ Qh, bf16* __restrict__ Kh, bf16* __restrict__ Vh)
{
  const int lane = threadIdx.x & 31;
  const int gw   = blockIdx.x * 8 + (threadIdx.x >> 5); // 0..61439
  const int t = gw / 24;
  const int h = gw % 24;
  const int d0 = lane * 4;

  const float* row = qkv + (size_t)t * 9216 + h * 128;
  float4 q = *(const float4*)(row + d0);
  float4 k = *(const float4*)(row + 3072 + d0);
  float4 v = *(const float4*)(row + 6144 + d0);

  float sq = q.x*q.x + q.y*q.y + q.z*q.z + q.w*q.w;
  float sk = k.x*k.x + k.y*k.y + k.z*k.z + k.w*k.w;
  #pragma unroll
  for (int m = 16; m >= 1; m >>= 1) {
    sq += __shfl_xor(sq, m, 32);
    sk += __shfl_xor(sk, m, 32);
  }
  const float rq = rsqrtf(sq * (1.f/128.f) + 1e-5f);
  const float rk = rsqrtf(sk * (1.f/128.f) + 1e-5f);

  const bool txt = (t < 512);
  const float* gqp = txt ? gaq : gq;
  const float* gkp = txt ? gak : gk;
  float4 g0 = *(const float4*)(gqp + d0);
  float4 g1 = *(const float4*)(gkp + d0);
  float4 c4 = *(const float4*)(rope_cos + (size_t)t * 128 + d0);
  float4 s4 = *(const float4*)(rope_sin + (size_t)t * 128 + d0);

  float q0 = q.x*rq*g0.x, q1 = q.y*rq*g0.y, q2 = q.z*rq*g0.z, q3 = q.w*rq*g0.w;
  float k0 = k.x*rk*g1.x, k1 = k.y*rk*g1.y, k2 = k.z*rk*g1.z, k3 = k.w*rk*g1.w;

  // RoPE: out[2i] = x[2i]*c[2i] - x[2i+1]*s[2i]; out[2i+1] = x[2i+1]*c[2i+1] + x[2i]*s[2i+1]
  const float scale = 0.08838834764831845f;  // 1/sqrt(128)
  float oq0 = (q0*c4.x - q1*s4.x) * scale;
  float oq1 = (q1*c4.y + q0*s4.y) * scale;
  float oq2 = (q2*c4.z - q3*s4.z) * scale;
  float oq3 = (q3*c4.w + q2*s4.w) * scale;
  float ok0 =  k0*c4.x - k1*s4.x;
  float ok1 =  k1*c4.y + k0*s4.y;
  float ok2 =  k2*c4.z - k3*s4.z;
  float ok3 =  k3*c4.w + k2*s4.w;

  const size_t out = ((size_t)h * 2560 + t) * 128 + d0;
  v4bf qo; qo[0]=f2bf(oq0); qo[1]=f2bf(oq1); qo[2]=f2bf(oq2); qo[3]=f2bf(oq3);
  v4bf ko; ko[0]=f2bf(ok0); ko[1]=f2bf(ok1); ko[2]=f2bf(ok2); ko[3]=f2bf(ok3);
  v4bf vo; vo[0]=f2bf(v.x); vo[1]=f2bf(v.y); vo[2]=f2bf(v.z); vo[3]=f2bf(v.w);
  *(v4bf*)(Qh + out) = qo;
  *(v4bf*)(Kh + out) = ko;
  *(v4bf*)(Vh + out) = vo;
}

// ---------------------------------------------------------------------------
// Flash attention. Grid (S/128, H), 256 threads (8 waves). Wave w owns query
// rows [w*16, w*16+16). Streams 20 key blocks of 128; online softmax; all
// matmuls via v_wmma_f32_16x16x32_bf16. Output f32 [S][INNER].
// ---------------------------------------------------------------------------
__global__ __launch_bounds__(256) void flash_attn_kernel(
    const bf16* __restrict__ Qh, const bf16* __restrict__ Kh,
    const bf16* __restrict__ Vh, float* __restrict__ Out)
{
  __shared__ __align__(16) bf16 Qs[128][136];      // [q][d]
  __shared__ __align__(16) bf16 Ks[128][136];      // [key][d]
  __shared__ __align__(16) bf16 Vt[128][136];      // [d][key] (transposed)
  __shared__ __align__(16) bf16 Ps[8][16][136];    // per-wave P staging [q][key]

  const int h  = blockIdx.y;
  const int qb = blockIdx.x;
  const int tid  = threadIdx.x;
  const int lane = tid & 31;
  const int w    = tid >> 5;
  const int half = lane >> 4;
  const int ln   = lane & 15;

  // Load Q block (128x128 bf16), 64 elems/thread as 8x b128.
  {
    const int r = tid >> 1;
    const int c = (tid & 1) * 64;
    const bf16* src = Qh + ((size_t)h * 2560 + (size_t)qb * 128 + r) * 128 + c;
    #pragma unroll
    for (int i = 0; i < 8; ++i)
      *(U128*)&Qs[r][c + i*8] = *(const U128*)(src + i*8);
  }
  __syncthreads();

  // Q fragments are loop-invariant: load once (4 WMMA K-steps of 32).
  v16bf qf[4];
  #pragma unroll
  for (int kd = 0; kd < 4; ++kd) {
    const bf16* p = &Qs[w*16 + ln][kd*32];
    qf[kd] = load_frag(p + half*16, p + half*16 + 8);
  }

  const v8f zero8 = {0.f,0.f,0.f,0.f,0.f,0.f,0.f,0.f};
  v8f o[8];
  #pragma unroll
  for (int i = 0; i < 8; ++i) o[i] = zero8;
  float mrow[8], lrow[8];
  #pragma unroll
  for (int r = 0; r < 8; ++r) { mrow[r] = -1e30f; lrow[r] = 0.f; }

  for (int kb = 0; kb < 20; ++kb) {
    __syncthreads();
    {
      // K block: straight b128 copy.
      const int r = tid >> 1;
      const int c = (tid & 1) * 64;
      const bf16* ks = Kh + ((size_t)h * 2560 + (size_t)kb * 128 + r) * 128 + c;
      #pragma unroll
      for (int i = 0; i < 8; ++i)
        *(U128*)&Ks[r][c + i*8] = *(const U128*)(ks + i*8);

      // V block transposed: thread owns row d of Vt, gathers 64 keys
      // (global b16 loads coalesce across lanes), packs to b128 LDS stores.
      const int vd  = tid & 127;
      const int vk0 = (tid >> 7) * 64;
      const bf16* vs = Vh + ((size_t)h * 2560 + (size_t)kb * 128 + vk0) * 128 + vd;
      #pragma unroll
      for (int i8 = 0; i8 < 8; ++i8) {
        unsigned short e[8];
        #pragma unroll
        for (int j = 0; j < 8; ++j)
          e[j] = __builtin_bit_cast(unsigned short, vs[(size_t)(i8*8 + j) * 128]);
        U128 p { (unsigned)e[0] | ((unsigned)e[1] << 16),
                 (unsigned)e[2] | ((unsigned)e[3] << 16),
                 (unsigned)e[4] | ((unsigned)e[5] << 16),
                 (unsigned)e[6] | ((unsigned)e[7] << 16) };
        *(U128*)&Vt[vd][vk0 + i8*8] = p;
      }

      if (kb + 1 < 20) {  // prefetch next key/value block (global_prefetch_b8)
        const size_t nb = ((size_t)h * 2560 + (size_t)(kb + 1) * 128 + (tid >> 1)) * 128
                          + (tid & 1) * 64;
        __builtin_prefetch(Kh + nb, 0, 3);
        __builtin_prefetch(Vh + nb, 0, 3);
      }
    }
    __syncthreads();

    // S = Q K^T  (K-dim = D = 128 -> 4 WMMA steps)
    v8f s[8];
    #pragma unroll
    for (int nt = 0; nt < 8; ++nt) {
      v8f sa = zero8;
      #pragma unroll
      for (int kd = 0; kd < 4; ++kd) {
        const bf16* bp = &Ks[nt*16 + ln][kd*32 + half*16];
        v16bf kf = load_frag(bp, bp + 8);
        sa = __builtin_amdgcn_wmma_f32_16x16x32_bf16(
            false, qf[kd], false, kf, (short)0, sa, false, false);
      }
      s[nt] = sa;
    }

    // Online softmax over this key block (rows m = half*8 + r).
    float mnew[8], alpha[8];
    #pragma unroll
    for (int r = 0; r < 8; ++r) {
      float mx = s[0][r];
      #pragma unroll
      for (int nt = 1; nt < 8; ++nt) mx = fmaxf(mx, s[nt][r]);
      mx = fmaxf(mx, __shfl_xor(mx, 1, 32));
      mx = fmaxf(mx, __shfl_xor(mx, 2, 32));
      mx = fmaxf(mx, __shfl_xor(mx, 4, 32));
      mx = fmaxf(mx, __shfl_xor(mx, 8, 32));
      mnew[r]  = fmaxf(mrow[r], mx);
      alpha[r] = __expf(mrow[r] - mnew[r]);
      mrow[r]  = mnew[r];
    }
    #pragma unroll
    for (int r = 0; r < 8; ++r) {
      float sum = 0.f;
      #pragma unroll
      for (int nt = 0; nt < 8; ++nt) {
        float p = __expf(s[nt][r] - mnew[r]);
        sum += p;
        Ps[w][half*8 + r][nt*16 + ln] = f2bf(p);
      }
      sum += __shfl_xor(sum, 1, 32);
      sum += __shfl_xor(sum, 2, 32);
      sum += __shfl_xor(sum, 4, 32);
      sum += __shfl_xor(sum, 8, 32);
      lrow[r] = lrow[r] * alpha[r] + sum;
      #pragma unroll
      for (int nt = 0; nt < 8; ++nt) o[nt][r] *= alpha[r];
    }

    // O += P V  (K-dim = 128 keys -> 4 WMMA steps)
    #pragma unroll
    for (int kc = 0; kc < 4; ++kc) {
      const bf16* ap = &Ps[w][ln][kc*32];
      v16bf pf = load_frag(ap + half*16, ap + half*16 + 8);
      #pragma unroll
      for (int nt = 0; nt < 8; ++nt) {
        const bf16* bp = &Vt[nt*16 + ln][kc*32 + half*16];
        v16bf vf = load_frag(bp, bp + 8);
        o[nt] = __builtin_amdgcn_wmma_f32_16x16x32_bf16(
            false, pf, false, vf, (short)0, o[nt], false, false);
      }
    }
  }

  // Finalize: O / l, write attn [t][h*128 + d] as f32.
  #pragma unroll
  for (int r = 0; r < 8; ++r) {
    const float inv = 1.f / lrow[r];
    const int row = qb*128 + w*16 + half*8 + r;
    float* dst = Out + (size_t)row * 3072 + h * 128;
    #pragma unroll
    for (int nt = 0; nt < 8; ++nt)
      dst[nt*16 + ln] = o[nt][r] * inv;
  }
}

// ---------------------------------------------------------------------------
extern "C" void kernel_launch(void* const* d_in, const int* in_sizes, int n_in,
                              void* d_out, int out_size, void* d_ws, size_t ws_size,
                              hipStream_t stream)
{
  const float* hidden   = (const float*)d_in[0];   // [1,2048,3072]
  const float* encoder  = (const float*)d_in[1];   // [1,512,3072]
  const float* rope_cos = (const float*)d_in[2];   // [2560,128]
  const float* rope_sin = (const float*)d_in[3];
  const float* Wqkv     = (const float*)d_in[4];   // [3072,9216]
  const float* Wqkv_add = (const float*)d_in[5];
  const float* b_add    = (const float*)d_in[6];
  const float* gq       = (const float*)d_in[7];
  const float* gk       = (const float*)d_in[8];
  const float* gaq      = (const float*)d_in[9];
  const float* gak      = (const float*)d_in[10];
  const float* Wout     = (const float*)d_in[11];  // [3072,3072]
  const float* bout     = (const float*)d_in[12];
  const float* Wadd_out = (const float*)d_in[13];
  const float* badd_out = (const float*)d_in[14];

  char* ws = (char*)d_ws;
  constexpr size_t QKV_BYTES  = (size_t)2560 * 9216 * 4;          // 94,371,840
  constexpr size_t HEAD_BYTES = (size_t)24 * 2560 * 128 * 2;      // 15,728,640
  float* qkv  = (float*)(ws);                                     // [2560][9216] txt first
  bf16*  Qh   = (bf16*)(ws + QKV_BYTES);
  bf16*  Kh   = (bf16*)(ws + QKV_BYTES + HEAD_BYTES);
  bf16*  Vh   = (bf16*)(ws + QKV_BYTES + 2 * HEAD_BYTES);
  float* attn = (float*)(ws + QKV_BYTES + 3 * HEAD_BYTES);        // [2560][3072]

  const dim3 blk(256);

  // 1) QKV projections (txt rows 0..511 with bias, img rows 512..2559).
  gemm_bf16_kernel<false><<<dim3(72, 16), blk, 0, stream>>>(
      hidden, Wqkv, nullptr, qkv + (size_t)512 * 9216, 2048, 9216, 3072);
  gemm_bf16_kernel<true><<<dim3(72, 4), blk, 0, stream>>>(
      encoder, Wqkv_add, b_add, qkv, 512, 9216, 3072);

  // 2) RMS-norm + RoPE + pack to bf16 [H][S][D].
  norm_rope_pack_kernel<<<7680, blk, 0, stream>>>(
      qkv, gq, gk, gaq, gak, rope_cos, rope_sin, Qh, Kh, Vh);

  // 3) Flash attention -> attn [2560][3072] f32.
  flash_attn_kernel<<<dim3(20, 24), blk, 0, stream>>>(Qh, Kh, Vh, attn);

  // 4) Output projections: img first, then ctx (matches d_out concat order).
  gemm_bf16_kernel<true><<<dim3(24, 16), blk, 0, stream>>>(
      attn + (size_t)512 * 3072, Wout, bout, (float*)d_out, 2048, 3072, 3072);
  gemm_bf16_kernel<true><<<dim3(24, 4), blk, 0, stream>>>(
      attn, Wadd_out, badd_out, (float*)d_out + (size_t)2048 * 3072, 512, 3072, 3072);
}
